// CommodityS4Model_85830626443733
// MI455X (gfx1250) — compile-verified, matrix operand order checked by reading, not run on hardware
//
#include <hip/hip_runtime.h>
#include <cstddef>
#include <cstdint>

// ---- problem sizes ----
#define B_SZ 64
#define L_SZ 2048
#define DIN  5
#define H_SZ 512
#define N_ST 64
#define NLAYERS 4
#define DEMB 128
#define KPAD 32                 // leading zeros => causal masking for Toeplitz tiles
#define KROW (L_SZ + KPAD)      // 2080

#ifndef __has_builtin
#define __has_builtin(x) 0
#endif
#if __has_builtin(__builtin_amdgcn_tensor_load_to_lds) && \
    __has_builtin(__builtin_amdgcn_s_wait_tensorcnt)
#define S4_HAS_TDM 1
#else
#define S4_HAS_TDM 0
#endif

typedef _Float16 v16h __attribute__((ext_vector_type(16)));
typedef _Float16 v8h  __attribute__((ext_vector_type(8)));
typedef float    v8f  __attribute__((ext_vector_type(8)));
typedef unsigned int v4u __attribute__((ext_vector_type(4)));
typedef int          v8i __attribute__((ext_vector_type(8)));
typedef int          v4i __attribute__((ext_vector_type(4)));

__device__ __forceinline__ v16h pack16(v8h lo, v8h hi) {
  v16h r;
#pragma unroll
  for (int i = 0; i < 8; ++i) { r[i] = lo[i]; r[8 + i] = hi[i]; }
  return r;
}

__device__ __forceinline__ float gelu_exact(float x) {
  return 0.5f * x * (1.f + erff(x * 0.70710678118654752f));
}

#if S4_HAS_TDM
// Build a D# for a 2D tile of 2-byte elements and fire TENSOR_LOAD_TO_LDS
// (6-arg toolchain form: g0, g1, g2, g3, extra group, cpol).
// Optional LDS padding: after every (2<<pad_interval) DWORDs stored, skip
// (pad_amount+1) DWORDs (hardware bank-conflict padding).
__device__ __forceinline__ void tdm_load_2d(uint32_t lds_addr, uint64_t gaddr,
                                            uint32_t tensor_w, uint32_t tensor_h,
                                            uint32_t stride_elems,
                                            uint32_t tile_w, uint32_t tile_h,
                                            uint32_t pad_enable,
                                            uint32_t pad_interval,
                                            uint32_t pad_amount) {
  v4u g0;
  g0[0] = 1u;                                            // count=1, user-mode D#
  g0[1] = lds_addr;                                      // lds_addr [63:32]
  g0[2] = (uint32_t)(gaddr & 0xffffffffu);               // global_addr[31:0]
  g0[3] = (uint32_t)((gaddr >> 32) & 0x01ffffffu) | (2u << 30);  // [120:96], type=2
  v8i g1;
  g1[0] = (int)((1u << 16) | (pad_enable << 20) | (pad_interval << 22) |
                (pad_amount << 25));                     // data_size=1 (2B), mask=0
  g1[1] = (int)((tensor_w & 0xffffu) << 16);             // tensor_dim0[15:0]
  g1[2] = (int)(((tensor_w >> 16) & 0xffffu) | ((tensor_h & 0xffffu) << 16));
  g1[3] = (int)(((tensor_h >> 16) & 0xffffu) | ((tile_w & 0xffffu) << 16));
  g1[4] = (int)(tile_h & 0xffffu);                       // tile_dim1, tile_dim2=0
  g1[5] = (int)stride_elems;                             // tensor_dim0_stride[31:0]
  g1[6] = 0;
  g1[7] = 0;
  v4i g2 = {0, 0, 0, 0};
  v4i g3 = {0, 0, 0, 0};
  v8i gx = {0, 0, 0, 0, 0, 0, 0, 0};
  __builtin_amdgcn_tensor_load_to_lds(g0, g1, g2, g3, gx, 0);
}
#endif

// ---------------- encoder: X[b,l,h] = seq[b,l,:5] @ enc_W + enc_b ----------------
__global__ void __launch_bounds__(256) s4_enc(const float* __restrict__ seq,
                                              const float* __restrict__ W,
                                              const float* __restrict__ bias,
                                              float* __restrict__ X) {
  int m = blockIdx.x;                       // b*L + l
  const float* s = seq + (size_t)m * DIN;
  float s0 = s[0], s1 = s[1], s2 = s[2], s3 = s[3], s4 = s[4];
  for (int h = threadIdx.x; h < H_SZ; h += 256) {
    float acc = bias[h] + s0 * W[h] + s1 * W[H_SZ + h] + s2 * W[2 * H_SZ + h] +
                s3 * W[3 * H_SZ + h] + s4 * W[4 * H_SZ + h];
    X[(size_t)m * H_SZ + h] = acc;
  }
}

// ---------------- fp32 -> fp16 ----------------
__global__ void __launch_bounds__(256) s4_tohalf(const float* __restrict__ src,
                                                 _Float16* __restrict__ dst, int n) {
  int i = blockIdx.x * 256 + threadIdx.x;
  if (i < n) dst[i] = (_Float16)src[i];
}

// ---------------- per-(h,n) discretization: Wb = (Re dtA, Im dtA, Re 2*C*dB, Im 2*C*dB) ----
__global__ void __launch_bounds__(256) s4_prep(const float* __restrict__ lstep,
                                               const float* __restrict__ Ar_,
                                               const float* __restrict__ Ai_,
                                               const float* __restrict__ Br_,
                                               const float* __restrict__ Bi_,
                                               const float* __restrict__ Cp_,
                                               float* __restrict__ Wb) {
  int idx = blockIdx.x * 256 + threadIdx.x;
  if (idx >= H_SZ * N_ST) return;
  int h = idx >> 6;
  float dt = __expf(lstep[h]);
  float Ar = -__expf(Ar_[idx]);
  float Ai = Ai_[idx];
  float ar = dt * Ar, ai = dt * Ai;          // dtA
  float e = __expf(ar), sn, cs;
  __sincosf(ai, &sn, &cs);
  float dAr = e * cs, dAi = e * sn;          // dA = exp(dtA)
  float nr = dAr - 1.f, ni = dAi;            // dA - 1
  float den = 1.f / (Ar * Ar + Ai * Ai);
  float tr = (nr * Ar + ni * Ai) * den;      // (dA-1)/A
  float ti = (ni * Ar - nr * Ai) * den;
  float Br = Br_[idx], Bi = Bi_[idx];
  float dBr = dt * (Br * tr - Bi * ti);
  float dBi = dt * (Br * ti + Bi * tr);
  float Cr = Cp_[idx * 2], Ci = Cp_[idx * 2 + 1];
  float wr = 2.f * (Cr * dBr - Ci * dBi);
  float wi = 2.f * (Cr * dBi + Ci * dBr);
  float4 o = make_float4(ar, ai, wr, wi);
  *(float4*)(Wb + (size_t)idx * 4) = o;
}

// ---------------- Vandermonde: kpad[h, KPAD+l] = Re(sum_n w*exp(dtA*l)), fp16 out ------
__global__ void __launch_bounds__(256) s4_vander(const float* __restrict__ Wb,
                                                 _Float16* __restrict__ kpad) {
  int h = blockIdx.x;
  int l = blockIdx.y * 256 + threadIdx.x;
  float fl = (float)l;
  const float* w = Wb + (size_t)h * N_ST * 4;
  float acc = 0.f;
#pragma unroll 4
  for (int n = 0; n < N_ST; ++n) {
    float ar = w[n * 4 + 0], ai = w[n * 4 + 1];
    float wr = w[n * 4 + 2], wi = w[n * 4 + 3];
    float e = __expf(ar * fl);
    float sn, cs;
    __sincosf(ai * fl, &sn, &cs);
    acc += e * (wr * cs - wi * sn);
  }
  kpad[(size_t)h * KROW + KPAD + l] = (_Float16)acc;
  if (l < KPAD) kpad[(size_t)h * KROW + l] = (_Float16)0.f;   // causal zero pad
}

// ---------------- LayerNorm over H, output transposed fp16 Zh[h][b][l] ----------------
__global__ void __launch_bounds__(256) s4_ln(const float* __restrict__ X,
                                             const float* __restrict__ g,
                                             const float* __restrict__ bt,
                                             _Float16* __restrict__ Zh) {
  int m = blockIdx.x;                 // b*L + l
  int b = m >> 11, l = m & (L_SZ - 1);
  const float* row = X + (size_t)m * H_SZ;
  int t = threadIdx.x;
  float x0 = row[t], x1 = row[t + 256];
  __shared__ float red[256];
  red[t] = x0 + x1;
  __syncthreads();
  for (int o = 128; o > 0; o >>= 1) { if (t < o) red[t] += red[t + o]; __syncthreads(); }
  float mu = red[0] * (1.f / H_SZ);
  __syncthreads();
  float d0 = x0 - mu, d1 = x1 - mu;
  red[t] = d0 * d0 + d1 * d1;
  __syncthreads();
  for (int o = 128; o > 0; o >>= 1) { if (t < o) red[t] += red[t + o]; __syncthreads(); }
  float rstd = rsqrtf(red[0] * (1.f / H_SZ) + 1e-5f);
  float z0 = d0 * rstd * g[t] + bt[t];
  float z1 = d1 * rstd * g[t + 256] + bt[t + 256];
  Zh[((size_t)t * B_SZ + b) * L_SZ + l] = (_Float16)z0;
  Zh[((size_t)(t + 256) * B_SZ + b) * L_SZ + l] = (_Float16)z1;
}

// ---------------- causal conv as block-Toeplitz WMMA GEMM, fused D*u + GELU --------------
// Per h: Y(b,l) = sum_t Z(b,t) * k[l-t];  A = Z tile (16b x 32t), B = Toeplitz (32t x 16l)
__global__ void __launch_bounds__(256) s4_conv(const _Float16* __restrict__ Zh,
                                               const _Float16* __restrict__ kpad,
                                               const float* __restrict__ Dp,
                                               _Float16* __restrict__ G) {
  int h = blockIdx.x;
  int nPair = blockIdx.y;
  __shared__ __align__(16) _Float16 kh[KROW];
#if S4_HAS_TDM
  if (threadIdx.x < 32) {
    tdm_load_2d((uint32_t)(size_t)&kh[0],
                (uint64_t)(size_t)(kpad + (size_t)h * KROW),
                KROW, 1, KROW, KROW, 1, /*pad*/0, 0, 0);
    __builtin_amdgcn_s_wait_tensorcnt(0);
  }
#else
  {
    const _Float16* kp = kpad + (size_t)h * KROW;
    for (int i = threadIdx.x; i < KROW; i += 256) kh[i] = kp[i];
  }
#endif
  __syncthreads();

  int wv = threadIdx.x >> 5;
  int lane = threadIdx.x & 31;
  int gg = lane >> 4;                 // lane-half selects K sub-ranges per ISA layout
  int n = lane & 15;                  // A row / B col / D col
  int mTile = wv & 3;
  int nTile = nPair * 2 + (wv >> 2);
  int bBase = mTile * 16;
  int nBase = nTile * 16;

  const _Float16* Arow = Zh + ((size_t)h * B_SZ + bBase + n) * L_SZ;
  v8f acc = {};
  for (int t0 = 0; t0 <= nBase; t0 += 32) {
    if (t0 + 32 <= nBase) __builtin_prefetch(Arow + t0 + 32, 0, 3);
    // A fragment: lanes 0-15 K={0..7,16..23}, lanes 16-31 K={8..15,24..31}
    const _Float16* ap = Arow + t0 + gg * 8;
    v8h alo = *(const v8h*)ap;
    v8h ahi = *(const v8h*)(ap + 16);
    v16h a = pack16(alo, ahi);
    // B fragment (Toeplitz): B[t][l] = kh[KPAD + l - t]; pad zeros enforce causality
    int base = KPAD + (nBase + n) - t0 - gg * 8;
    v16h bf;
#pragma unroll
    for (int i = 0; i < 8; ++i) { bf[i] = kh[base - i]; bf[8 + i] = kh[base - 16 - i]; }
    acc = __builtin_amdgcn_wmma_f32_16x16x32_f16(false, a, false, bf, (short)0, acc,
                                                 false, false);
  }
  // epilogue: y + D*u, exact GELU, write fp16 into (B*L, H) GEMM A-matrix
  float Dh = Dp[h];
  int l = nBase + n;
#pragma unroll
  for (int j = 0; j < 8; ++j) {
    int b = bBase + j + 8 * gg;       // D layout: row = vgpr + 8*(lane>=16)
    float z = (float)Zh[((size_t)h * B_SZ + b) * L_SZ + l];
    float y = acc[j] + Dh * z;
    G[((size_t)b * L_SZ + l) * H_SZ + h] = (_Float16)gelu_exact(y);
  }
}

// ---------------- out_W GEMM: X += G(131072x512,f16) @ W(512x512,f16) + bias -------------
// A-panels arrive via TDM (double buffered, LDS pad 4 DW per 16 DW -> 40-half rows);
// B-panels are transposed by the block while the TDM is in flight.
__global__ void __launch_bounds__(256) s4_gemm(const _Float16* __restrict__ G,
                                               const _Float16* __restrict__ Wl,
                                               const float* __restrict__ bias,
                                               float* __restrict__ X) {
  __shared__ __align__(16) _Float16 As[2][128][40];
  __shared__ __align__(16) _Float16 Bs[2][64][40];
  int mBase = blockIdx.x * 128;
  int nBase = blockIdx.y * 64;
  int tid = threadIdx.x;
  int wv = tid >> 5, lane = tid & 31, gg = lane >> 4, n = lane & 15;
  v8f acc[4] = {{}, {}, {}, {}};

  // stage k-block 0 into buffer 0
#if S4_HAS_TDM
  if (tid < 32) {
    tdm_load_2d((uint32_t)(size_t)&As[0][0][0],
                (uint64_t)(size_t)(G + (size_t)mBase * H_SZ),
                H_SZ, 1u << 20, H_SZ, /*tile_w*/32, /*tile_h*/128,
                /*pad*/1, /*interval 16DW*/3, /*amount 4DW*/3);
  }
#else
  {
    int r = tid & 127, seg = tid >> 7;
    const _Float16* p = G + (size_t)(mBase + r) * H_SZ + seg * 16;
    *(v8h*)&As[0][r][seg * 16] = *(const v8h*)p;
    *(v8h*)&As[0][r][seg * 16 + 8] = *(const v8h*)(p + 8);
  }
#endif
  {
    int nn = tid & 63, kq = tid >> 6;
#pragma unroll
    for (int i = 0; i < 8; ++i)
      Bs[0][nn][kq * 8 + i] = Wl[(size_t)(kq * 8 + i) * H_SZ + nBase + nn];
  }
#if S4_HAS_TDM
  if (tid < 32) __builtin_amdgcn_s_wait_tensorcnt(0);
#endif
  __syncthreads();

  for (int kb = 0; kb < H_SZ; kb += 32) {
    int cur = (kb >> 5) & 1;
    int nxt = cur ^ 1;
    bool more = (kb + 32) < H_SZ;
    if (more) {
      // issue next A-panel DMA, then transpose next B-panel while it flies
#if S4_HAS_TDM
      if (tid < 32) {
        tdm_load_2d((uint32_t)(size_t)&As[nxt][0][0],
                    (uint64_t)(size_t)(G + (size_t)mBase * H_SZ + kb + 32),
                    H_SZ, 1u << 20, H_SZ, 32, 128, 1, 3, 3);
      }
#else
      {
        int r = tid & 127, seg = tid >> 7;
        const _Float16* p = G + (size_t)(mBase + r) * H_SZ + kb + 32 + seg * 16;
        *(v8h*)&As[nxt][r][seg * 16] = *(const v8h*)p;
        *(v8h*)&As[nxt][r][seg * 16 + 8] = *(const v8h*)(p + 8);
      }
#endif
      int nn = tid & 63, kq = tid >> 6;
#pragma unroll
      for (int i = 0; i < 8; ++i)
        Bs[nxt][nn][kq * 8 + i] =
            Wl[(size_t)(kb + 32 + kq * 8 + i) * H_SZ + nBase + nn];
    }
    // compute on current buffer
    int arow = wv * 16 + n;
    v8h alo = *(const v8h*)&As[cur][arow][gg * 8];
    v8h ahi = *(const v8h*)&As[cur][arow][gg * 8 + 16];
    v16h a = pack16(alo, ahi);
#pragma unroll
    for (int nt = 0; nt < 4; ++nt) {
      int col = nt * 16 + n;
      v8h blo = *(const v8h*)&Bs[cur][col][gg * 8];
      v8h bhi = *(const v8h*)&Bs[cur][col][gg * 8 + 16];
      v16h bf = pack16(blo, bhi);
      acc[nt] = __builtin_amdgcn_wmma_f32_16x16x32_f16(false, a, false, bf, (short)0,
                                                       acc[nt], false, false);
    }
    if (more) {
#if S4_HAS_TDM
      if (tid < 32) __builtin_amdgcn_s_wait_tensorcnt(0);
#endif
      __syncthreads();
    }
  }
  // epilogue: bias + residual, fp32 store
#pragma unroll
  for (int nt = 0; nt < 4; ++nt) {
    int col = nBase + nt * 16 + n;
    float bb = bias[col];
#pragma unroll
    for (int j = 0; j < 8; ++j) {
      int m = mBase + wv * 16 + j + 8 * gg;
      size_t idx = (size_t)m * H_SZ + col;
      X[idx] = X[idx] + acc[nt][j] + bb;
    }
  }
}

// ---------------- mean pool over L ----------------
__global__ void __launch_bounds__(256) s4_pool(const float* __restrict__ X,
                                               float* __restrict__ pool) {
  int b = blockIdx.x, t = threadIdx.x;
  const float* base = X + (size_t)b * L_SZ * H_SZ;
  float a0 = 0.f, a1 = 0.f;
  for (int l = 0; l < L_SZ; ++l) {
    a0 += base[(size_t)l * H_SZ + t];
    a1 += base[(size_t)l * H_SZ + t + 256];
  }
  pool[b * H_SZ + t] = a0 * (1.f / L_SZ);
  pool[b * H_SZ + t + 256] = a1 * (1.f / L_SZ);
}

// ---------------- fused head: concat -> fusion -> two ReLU MLPs -> (B,2) ----------------
__global__ void __launch_bounds__(256) s4_head(const float* __restrict__ pool,
                                               const int* __restrict__ com,
                                               const int* __restrict__ cou,
                                               const float* __restrict__ com_emb,
                                               const float* __restrict__ cou_emb,
                                               const float* __restrict__ fW,
                                               const float* __restrict__ fb,
                                               const float* __restrict__ vW1,
                                               const float* __restrict__ vb1,
                                               const float* __restrict__ vW2,
                                               const float* __restrict__ vb2,
                                               const float* __restrict__ wW1,
                                               const float* __restrict__ wb1,
                                               const float* __restrict__ wW2,
                                               const float* __restrict__ wb2,
                                               float* __restrict__ out) {
  int b = blockIdx.x, t = threadIdx.x;
  __shared__ float feat[H_SZ + 2 * DEMB];
  __shared__ float hbuf[H_SZ];
  __shared__ float red[256];
  feat[t] = pool[b * H_SZ + t];
  feat[t + 256] = pool[b * H_SZ + t + 256];
  if (t < DEMB) {
    feat[H_SZ + t] = com_emb[com[b] * DEMB + t];
    feat[H_SZ + DEMB + t] = cou_emb[cou[b] * DEMB + t];
  }
  __syncthreads();
  for (int rep = 0; rep < 2; ++rep) {
    int nn = t + rep * 256;
    float acc = fb[nn];
    for (int k = 0; k < H_SZ + 2 * DEMB; ++k) acc += feat[k] * fW[(size_t)k * H_SZ + nn];
    hbuf[nn] = acc;
  }
  __syncthreads();
  float av = vb1[t], aw = wb1[t];
  for (int k = 0; k < H_SZ; ++k) {
    float hv = hbuf[k];
    av += hv * vW1[(size_t)k * 256 + t];
    aw += hv * wW1[(size_t)k * 256 + t];
  }
  av = fmaxf(av, 0.f) * vW2[t];
  aw = fmaxf(aw, 0.f) * wW2[t];
  red[t] = av;
  __syncthreads();
  for (int o = 128; o > 0; o >>= 1) { if (t < o) red[t] += red[t + o]; __syncthreads(); }
  if (t == 0) out[b * 2 + 0] = red[0] + vb2[0];
  __syncthreads();
  red[t] = aw;
  __syncthreads();
  for (int o = 128; o > 0; o >>= 1) { if (t < o) red[t] += red[t + o]; __syncthreads(); }
  if (t == 0) out[b * 2 + 1] = red[0] + wb2[0];
}

// ================================ host launcher ================================
extern "C" void kernel_launch(void* const* d_in, const int* in_sizes, int n_in,
                              void* d_out, int out_size, void* d_ws, size_t ws_size,
                              hipStream_t stream) {
  const float* seq      = (const float*)d_in[0];
  const int*   com      = (const int*)d_in[1];
  const int*   cou      = (const int*)d_in[2];
  const float* enc_W    = (const float*)d_in[3];
  const float* enc_b    = (const float*)d_in[4];
  const float* com_emb  = (const float*)d_in[5];
  const float* cou_emb  = (const float*)d_in[6];
  const float* log_step = (const float*)d_in[7];
  const float* A_real   = (const float*)d_in[8];
  const float* A_imag   = (const float*)d_in[9];
  const float* B_real   = (const float*)d_in[10];
  const float* B_imag   = (const float*)d_in[11];
  const float* C_param  = (const float*)d_in[12];
  const float* D_param  = (const float*)d_in[13];
  const float* out_W    = (const float*)d_in[14];
  const float* out_b    = (const float*)d_in[15];
  const float* ln_g     = (const float*)d_in[16];
  const float* ln_b     = (const float*)d_in[17];
  const float* fusion_W = (const float*)d_in[18];
  const float* fusion_b = (const float*)d_in[19];
  const float* val_W1   = (const float*)d_in[20];
  const float* val_b1   = (const float*)d_in[21];
  const float* val_W2   = (const float*)d_in[22];
  const float* val_b2   = (const float*)d_in[23];
  const float* wt_W1    = (const float*)d_in[24];
  const float* wt_b1    = (const float*)d_in[25];
  const float* wt_W2    = (const float*)d_in[26];
  const float* wt_b2    = (const float*)d_in[27];

  char* ws = (char*)d_ws;
  size_t off = 0;
  float*    X   = (float*)(ws + off);    off += (size_t)B_SZ * L_SZ * H_SZ * 4;   // 256 MB
  _Float16* Zh  = (_Float16*)(ws + off); off += (size_t)H_SZ * B_SZ * L_SZ * 2;   // 128 MB
  _Float16* G   = (_Float16*)(ws + off); off += (size_t)B_SZ * L_SZ * H_SZ * 2;   // 128 MB
  _Float16* Kp  = (_Float16*)(ws + off); off += (size_t)H_SZ * KROW * 2;
  float*    Wb  = (float*)(ws + off);    off += (size_t)H_SZ * N_ST * 4 * 4;
  _Float16* W16 = (_Float16*)(ws + off); off += (size_t)NLAYERS * H_SZ * H_SZ * 2;
  float*    pl  = (float*)(ws + off);    off += (size_t)B_SZ * H_SZ * 4;

  s4_enc<<<dim3(B_SZ * L_SZ), 256, 0, stream>>>(seq, enc_W, enc_b, X);
  s4_tohalf<<<dim3((NLAYERS * H_SZ * H_SZ) / 256), 256, 0, stream>>>(
      out_W, W16, NLAYERS * H_SZ * H_SZ);

  for (int i = 0; i < NLAYERS; ++i) {
    s4_prep<<<dim3(H_SZ * N_ST / 256), 256, 0, stream>>>(
        log_step + (size_t)i * H_SZ,
        A_real + (size_t)i * H_SZ * N_ST, A_imag + (size_t)i * H_SZ * N_ST,
        B_real + (size_t)i * H_SZ * N_ST, B_imag + (size_t)i * H_SZ * N_ST,
        C_param + (size_t)i * H_SZ * N_ST * 2, Wb);
    s4_vander<<<dim3(H_SZ, L_SZ / 256), 256, 0, stream>>>(Wb, Kp);
    s4_ln<<<dim3(B_SZ * L_SZ), 256, 0, stream>>>(X, ln_g + (size_t)i * H_SZ,
                                                 ln_b + (size_t)i * H_SZ, Zh);
    s4_conv<<<dim3(H_SZ, L_SZ / 32), 256, 0, stream>>>(Zh, Kp,
                                                       D_param + (size_t)i * H_SZ, G);
    s4_gemm<<<dim3(B_SZ * L_SZ / 128, H_SZ / 64), 256, 0, stream>>>(
        G, W16 + (size_t)i * H_SZ * H_SZ, out_b + (size_t)i * H_SZ, X);
  }

  s4_pool<<<dim3(B_SZ), 256, 0, stream>>>(X, pl);
  s4_head<<<dim3(B_SZ), 256, 0, stream>>>(pl, com, cou, com_emb, cou_emb,
                                          fusion_W, fusion_b, val_W1, val_b1, val_W2,
                                          val_b2, wt_W1, wt_b1, wt_W2, wt_b2,
                                          (float*)d_out);
}
